// ImplicitFrequencyCNN_23330262351890
// MI455X (gfx1250) — compile-verified, matrix operand order checked by reading, not run on hardware
//
#include <hip/hip_runtime.h>
#include <hip/hip_bf16.h>
#include <math.h>

// Problem constants (from reference)
#define BATCH   8
#define C_IN    32
#define C_OUT   32
#define TLEN    8192
#define TLOG2   13
#define HID     64
#define FDIM    (C_IN * C_OUT)   // 1024

typedef __attribute__((ext_vector_type(16))) _Float16 v16h;
typedef __attribute__((ext_vector_type(8)))  float    v8f;
typedef int v2i __attribute__((__vector_size__(8)));
typedef int v4i __attribute__((__vector_size__(16)));

#define AS1 __attribute__((address_space(1)))
#define AS3 __attribute__((address_space(3)))

// gfx1250 async global<->LDS paths (guarded: fall back to plain ds stores)
#if defined(__gfx1250__) && __has_builtin(__builtin_amdgcn_global_load_async_to_lds_b32)
#define HAVE_ASYNC_LD32 1
#endif
#if defined(__gfx1250__) && __has_builtin(__builtin_amdgcn_global_load_async_to_lds_b64)
#define HAVE_ASYNC_LD64 1
#endif
#if defined(__gfx1250__) && __has_builtin(__builtin_amdgcn_global_store_async_from_lds_b128)
#define HAVE_ASYNC_ST128 1
#endif

__device__ __forceinline__ void wait_async_all() {
#if __has_builtin(__builtin_amdgcn_s_wait_asynccnt)
  __builtin_amdgcn_s_wait_asynccnt(0);
#else
  asm volatile("s_wait_asynccnt 0" ::: "memory");
#endif
}

// ---------------------------------------------------------------------------
// Helpers
// ---------------------------------------------------------------------------
__device__ __forceinline__ float gelu_exact(float x) {
  return 0.5f * x * (1.0f + erff(x * 0.70710678118654752440f));
}

__device__ __forceinline__ float2 cmul(float2 a, float2 b) {
  return make_float2(a.x * b.x - a.y * b.y, a.x * b.y + a.y * b.x);
}

// ---------------------------------------------------------------------------
// Kernel 1: implicit filter MLP via WMMA (f16 inputs, f32 accumulate)
//   h1 = gelu(pos*w1 + b1)            (16 x 64)   elementwise (fan_in = 1)
//   h2 = gelu(h1 @ w2 + b2)           (16x64)@(64x64)    -> WMMA
//   f  = h2 @ w3 + b3                 (16x64)@(64x1024)  -> WMMA
//   filt[c][t] = f[t][c]   (c = o*32+i is already the (o,i) filter row index)
//
// One workgroup = 16 rows of t (M=16), 128 threads = 4 waves (wave32).
// Fragment layouts per CDNA5 ISA 7.12.2.
// ---------------------------------------------------------------------------
__global__ void mlp_filters_kernel(const float* __restrict__ w1,
                                   const float* __restrict__ b1,
                                   const float* __restrict__ w2,
                                   const float* __restrict__ b2,
                                   const float* __restrict__ w3,
                                   const float* __restrict__ b3,
                                   float* __restrict__ filt /* (FDIM, TLEN) */) {
  __shared__ _Float16 h1[16 * HID];
  __shared__ _Float16 h2[16 * HID];

  const int tbase = blockIdx.x * 16;
  const int tid   = threadIdx.x;      // 128 threads
  const int wave  = tid >> 5;         // 4 waves
  const int lane  = tid & 31;

  // Phase A: h1 = gelu(pos * w1 + b1), pos[t] = t/(T-1)
  for (int e = tid; e < 16 * HID; e += blockDim.x) {
    const int r = e / HID, j = e % HID;
    const float p = (float)(tbase + r) * (1.0f / (float)(TLEN - 1));
    h1[e] = (_Float16)gelu_exact(p * w1[j] + b1[j]);
  }
  __syncthreads();

  const int mn    = lane & 15;
  const int khalf = (lane < 16) ? 0 : 8;
  const int mrow  = (lane < 16) ? 0 : 8;

  // Phase B: h2 = gelu(h1 @ w2 + b2). Wave w owns N-tile [16w, 16w+16).
  {
    const int cb = wave * 16;
    v8f acc = {};
    #pragma unroll
    for (int ks = 0; ks < 2; ++ks) {
      const int k0 = ks * 32;
      v16h a, bf;
      #pragma unroll
      for (int v = 0; v < 8; ++v) {
        const int kb = k0 + ((v < 4) ? 0 : 16) + khalf + 2 * (v & 3);
        a[2 * v]     = h1[mn * HID + kb];
        a[2 * v + 1] = h1[mn * HID + kb + 1];
        bf[2 * v]     = (_Float16)w2[kb * HID + cb + mn];
        bf[2 * v + 1] = (_Float16)w2[(kb + 1) * HID + cb + mn];
      }
      acc = __builtin_amdgcn_wmma_f32_16x16x32_f16(
          false, a, false, bf, (short)0, acc, false, false);
    }
    const float bb = b2[cb + mn];
    #pragma unroll
    for (int v = 0; v < 8; ++v) {
      const int m = v + mrow;
      h2[m * HID + cb + mn] = (_Float16)gelu_exact(acc[v] + bb);
    }
  }
  __syncthreads();

  // Phase C: f = h2 @ w3 + b3 ; scatter-store transposed into filt (FDIM, T).
  #pragma unroll 2
  for (int nt = wave; nt < FDIM / 16; nt += 4) {
    const int cb = nt * 16;
    v8f acc = {};
    #pragma unroll
    for (int ks = 0; ks < 2; ++ks) {
      const int k0 = ks * 32;
      v16h a, bf;
      #pragma unroll
      for (int v = 0; v < 8; ++v) {
        const int kb = k0 + ((v < 4) ? 0 : 16) + khalf + 2 * (v & 3);
        a[2 * v]     = h2[mn * HID + kb];
        a[2 * v + 1] = h2[mn * HID + kb + 1];
        bf[2 * v]     = (_Float16)w3[(size_t)kb * FDIM + cb + mn];
        bf[2 * v + 1] = (_Float16)w3[(size_t)(kb + 1) * FDIM + cb + mn];
      }
      acc = __builtin_amdgcn_wmma_f32_16x16x32_f16(
          false, a, false, bf, (short)0, acc, false, false);
    }
    const int   c  = cb + mn;        // column index == filter row (o*C_IN+i)
    const float bb = b3[c];
    #pragma unroll
    for (int v = 0; v < 8; ++v) {
      const int m = v + mrow;
      filt[(size_t)c * TLEN + (tbase + m)] = acc[v] + bb;
    }
  }
}

// ---------------------------------------------------------------------------
// In-place radix-2 DIT FFT on 8192 complex points resident in LDS.
// Data must be loaded in bit-reversed order. Twiddles come from a shared
// table tw[j] = exp(dir * i * 2*pi * j / N), j in [0, N/2):
//   w(stage,pos) = tw[pos << (TLOG2 - stage)]
// ---------------------------------------------------------------------------
__device__ void fft8192_lds(float2* s, const float2* tw, int tid, int nthreads) {
  for (int stage = 1; stage <= TLOG2; ++stage) {
    const int half  = 1 << (stage - 1);
    const int shift = TLOG2 - stage;
    for (int b = tid; b < (TLEN >> 1); b += nthreads) {
      const int grp = b >> (stage - 1);
      const int pos = b & (half - 1);
      const int i0  = (grp << stage) + pos;
      const int i1  = i0 + half;
      const float2 w = tw[pos << shift];
      const float2 u = s[i0];
      const float2 t = cmul(w, s[i1]);
      s[i0] = make_float2(u.x + t.x, u.y + t.y);
      s[i1] = make_float2(u.x - t.x, u.y - t.y);
    }
    __syncthreads();
  }
}

__device__ __forceinline__ void build_twiddles(float2* tw, int tid, int nthreads,
                                               float dir) {
  const float step = dir * 6.28318530717958647692f / (float)TLEN;
  for (int j = tid; j < (TLEN >> 1); j += nthreads) {
    float sn, cs;
    __sincosf(step * (float)j, &sn, &cs);
    tw[j] = make_float2(cs, sn);
  }
}

// Kernel 2/3: forward FFT of real rows -> full complex spectrum (one row / WG)
// LDS: [0, 64K) data, [64K, 96K) twiddle table.
__global__ void fft_rows_real_fwd(const float* __restrict__ in,
                                  float2* __restrict__ out) {
  extern __shared__ float2 smem[];
  float2* tw = smem + TLEN;
  const size_t row = blockIdx.x;
  const float* src = in + row * TLEN;

  // Zero the imaginary parts (plain ds stores).
  for (int i = threadIdx.x; i < TLEN; i += blockDim.x) smem[i].y = 0.0f;

#ifdef HAVE_ASYNC_LD32
  // Bit-reversed scatter straight into LDS via the async DMA path; the VALU
  // overlaps this with the twiddle-table sincos burst below.
  for (int i = threadIdx.x; i < TLEN; i += blockDim.x) {
    const unsigned r = __brev((unsigned)i) >> (32 - TLOG2);
    __builtin_amdgcn_global_load_async_to_lds_b32(
        (AS1 int*)(void*)(src + i), (AS3 int*)(void*)&smem[r].x, 0, 0);
  }
  build_twiddles(tw, threadIdx.x, blockDim.x, -1.0f);
  wait_async_all();
#else
  for (int i = threadIdx.x; i < TLEN; i += blockDim.x) {
    const unsigned r = __brev((unsigned)i) >> (32 - TLOG2);
    smem[r].x = src[i];
  }
  build_twiddles(tw, threadIdx.x, blockDim.x, -1.0f);
#endif
  __syncthreads();

  fft8192_lds(smem, tw, threadIdx.x, blockDim.x);

  float2* dst = out + row * TLEN;
#ifdef HAVE_ASYNC_ST128
  // Contiguous LDS -> global spill: 16B per async store.
  for (int i = threadIdx.x * 2; i < TLEN; i += blockDim.x * 2) {
    __builtin_amdgcn_global_store_async_from_lds_b128(
        (AS1 v4i*)(void*)(dst + i), (AS3 v4i*)(void*)&smem[i], 0, 0);
  }
  wait_async_all();
#else
  for (int i = threadIdx.x; i < TLEN; i += blockDim.x) dst[i] = smem[i];
#endif
}

// ---------------------------------------------------------------------------
// Kernel 4: per-bin complex channel contraction
//   Yf[b,o,f] = sum_i Xf[b,i,f] * Ff[o,i,f]
// Consecutive threads walk consecutive f -> fully coalesced; working set is
// L2-resident (96 MB), so this streams at cache bandwidth.
// ---------------------------------------------------------------------------
__global__ void freq_einsum_kernel(const float2* __restrict__ Xf,
                                   const float2* __restrict__ Ff,
                                   float2* __restrict__ Yf) {
  const size_t idx = (size_t)blockIdx.x * blockDim.x + threadIdx.x;
  const int f  = (int)(idx & (TLEN - 1));
  const int bo = (int)(idx >> TLOG2);
  const int o  = bo % C_OUT;
  const int b  = bo / C_OUT;
  const float2* xrow = Xf + (size_t)b * C_IN * TLEN + f;
  const float2* frow = Ff + (size_t)o * C_IN * TLEN + f;
  __builtin_prefetch((const void*)(xrow + (size_t)16 * TLEN), 0, 0);
  __builtin_prefetch((const void*)(frow + (size_t)16 * TLEN), 0, 0);
  float2 acc = make_float2(0.0f, 0.0f);
  #pragma unroll 4
  for (int i = 0; i < C_IN; ++i) {
    const float2 xv = xrow[(size_t)i * TLEN];
    const float2 fv = frow[(size_t)i * TLEN];
    acc.x = fmaf(xv.x, fv.x, fmaf(-xv.y, fv.y, acc.x));
    acc.y = fmaf(xv.x, fv.y, fmaf( xv.y, fv.x, acc.y));
  }
  Yf[idx] = acc;
}

// Kernel 5: inverse FFT per row, keep real part, scale 1/N, add bias[o]
__global__ void ifft_rows_real_out(const float2* __restrict__ in,
                                   const float* __restrict__ bias,
                                   float* __restrict__ out) {
  extern __shared__ float2 smem[];
  float2* tw = smem + TLEN;
  const size_t row = blockIdx.x;         // b*C_OUT + o
  const int o = (int)(row % C_OUT);
  const float2* src = in + row * TLEN;

#ifdef HAVE_ASYNC_LD64
  for (int i = threadIdx.x; i < TLEN; i += blockDim.x) {
    const unsigned r = __brev((unsigned)i) >> (32 - TLOG2);
    __builtin_amdgcn_global_load_async_to_lds_b64(
        (AS1 v2i*)(void*)(src + i), (AS3 v2i*)(void*)&smem[r], 0, 0);
  }
  build_twiddles(tw, threadIdx.x, blockDim.x, +1.0f);
  wait_async_all();
#else
  for (int i = threadIdx.x; i < TLEN; i += blockDim.x) {
    const unsigned r = __brev((unsigned)i) >> (32 - TLOG2);
    smem[r] = src[i];
  }
  build_twiddles(tw, threadIdx.x, blockDim.x, +1.0f);
#endif
  __syncthreads();

  fft8192_lds(smem, tw, threadIdx.x, blockDim.x);

  const float bo  = bias[o];
  const float inv = 1.0f / (float)TLEN;
  float* dst = out + row * TLEN;
  for (int i = threadIdx.x; i < TLEN; i += blockDim.x)
    dst[i] = smem[i].x * inv + bo;
}

// ---------------------------------------------------------------------------
// Launch: 5-stage pipeline. Workspace layout (bytes):
//   [0,   32M) filt  (FDIM x T, f32)
//   [32M, 96M) Ff    (FDIM x T, cf32)
//   [96M, 112M) Xf   (B*C_IN x T, cf32)
//   [112M,128M) Yf   (B*C_OUT x T, cf32)
// ---------------------------------------------------------------------------
extern "C" void kernel_launch(void* const* d_in, const int* in_sizes, int n_in,
                              void* d_out, int out_size, void* d_ws, size_t ws_size,
                              hipStream_t stream) {
  (void)in_sizes; (void)n_in; (void)out_size; (void)ws_size;

  const float* x    = (const float*)d_in[0];
  const float* w1   = (const float*)d_in[1];
  const float* b1   = (const float*)d_in[2];
  const float* w2   = (const float*)d_in[3];
  const float* b2   = (const float*)d_in[4];
  const float* w3   = (const float*)d_in[5];
  const float* b3   = (const float*)d_in[6];
  const float* bias = (const float*)d_in[7];
  float* out = (float*)d_out;

  char* ws = (char*)d_ws;
  float*  filt = (float*) (ws);
  float2* Ff   = (float2*)(ws + ((size_t)32  << 20));
  float2* Xf   = (float2*)(ws + ((size_t)96  << 20));
  float2* Yf   = (float2*)(ws + ((size_t)112 << 20));

  // 64 KB FFT data + 32 KB twiddle table per workgroup (320 KB LDS per WGP).
  const size_t fft_lds = sizeof(float2) * (TLEN + TLEN / 2);

  // 1) implicit filters via WMMA MLP
  mlp_filters_kernel<<<TLEN / 16, 128, 0, stream>>>(w1, b1, w2, b2, w3, b3, filt);
  // 2) spectra of filters (1024 rows)
  fft_rows_real_fwd<<<FDIM, 256, fft_lds, stream>>>(filt, Ff);
  // 3) spectra of x (256 rows)
  fft_rows_real_fwd<<<BATCH * C_IN, 256, fft_lds, stream>>>(x, Xf);
  // 4) per-bin complex channel contraction
  freq_einsum_kernel<<<(BATCH * C_OUT * TLEN) / 256, 256, 0, stream>>>(Xf, Ff, Yf);
  // 5) inverse FFT + bias
  ifft_rows_real_out<<<BATCH * C_OUT, 256, fft_lds, stream>>>(Yf, bias, out);
}
